// TensorizedGRU_68753836474916
// MI455X (gfx1250) — matrix operand order repaired and checked by reading, not compiled, not run on hardware
//
#include <hip/hip_runtime.h>
#include <cstdint>
#include <cstddef>

typedef __attribute__((ext_vector_type(16))) _Float16 v16h;
typedef __attribute__((ext_vector_type(8)))  float    v8f;
typedef __attribute__((ext_vector_type(4)))  unsigned int u32x4;
typedef __attribute__((ext_vector_type(8)))  int          i32x8;
typedef __attribute__((ext_vector_type(4)))  int          i32x4;

#define BSZ   4096
#define HH    128
#define TWOH  256            // 2H == 2I
#define KTOT  65536          // (2H)*(2I)
#define NFK   2048           // KTOT/32 K-fragments
#define NFN   8              // 128/16  N-fragments
#define FRAG  512            // halfs per 32x16 fragment

// workspace layout (in _Float16 elements)
#define X16_OFF  ((size_t)0)
#define S16_OFF  (X16_OFF + (size_t)BSZ * TWOH)
#define W1P_OFF  (S16_OFF + (size_t)BSZ * TWOH)
#define WP_SZ    ((size_t)NFK * NFN * FRAG)          // 8,388,608 halfs
#define W2P_OFF  (W1P_OFF + WP_SZ)
#define W3P_OFF  (W2P_OFF + WP_SZ)

union V16U { v16h h; uint4 q[2]; };

// Fragment layout: element (lane, i) at (i>>3)*256 + lane*8 + (i&7)
__device__ __forceinline__ v16h load_frag(const _Float16* p, int lane) {
  V16U t;
  t.q[0] = *(const uint4*)(p + lane * 8);
  t.q[1] = *(const uint4*)(p + 256 + lane * 8);
  return t.h;
}

// A-matrix 16x32 f16 per-lane row chunk (contiguous k = 16*h + i)
__device__ __forceinline__ v16h load_arow(const _Float16* rowp, int h) {
  V16U t;
  t.q[0] = *(const uint4*)(rowp + 16 * h);
  t.q[1] = *(const uint4*)(rowp + 16 * h + 8);
  return t.h;
}

__global__ void pack_sx(const float* __restrict__ in0, const float* __restrict__ in1,
                        const float* __restrict__ st0, const float* __restrict__ st1,
                        _Float16* __restrict__ ws) {
  int row = blockIdx.x, col = threadIdx.x;          // 4096 x 256
  float xv = (col < HH) ? in0[row * HH + col] : in1[row * HH + col - HH];
  float sv = (col < HH) ? st0[row * HH + col] : st1[row * HH + col - HH];
  ws[X16_OFF + (size_t)row * TWOH + col] = (_Float16)xv;
  ws[S16_OFF + (size_t)row * TWOH + col] = (_Float16)sv;
}

__global__ void pack_W(const float* __restrict__ W1, const float* __restrict__ W2,
                       _Float16* __restrict__ ws) {
  size_t tid = (size_t)blockIdx.x * 256 + threadIdx.x;   // 2^24 total
  int w = (int)(tid >> 23);
  const float* W = w ? W2 : W1;
  size_t r  = tid & ((1u << 23) - 1);
  int f   = (int)(r >> 9);
  int pos = (int)(r & 511);
  int kt = f >> 3, nt = f & 7;
  int half = pos >> 8, lane = (pos >> 3) & 31, io = pos & 7;
  int i = half * 8 + io;
  int K = ((lane >> 4) << 4) + i;
  int l = nt * 16 + (lane & 15);
  size_t jk = (size_t)kt * 32 + K;
  ws[(w ? W2P_OFF : W1P_OFF) + (size_t)f * FRAG + pos] =
      (_Float16)W[(size_t)l * KTOT + jk];
}

__global__ void pack_W3(const float* __restrict__ W3, _Float16* __restrict__ ws) {
  int tid = blockIdx.x * 256 + threadIdx.x;          // 64 frags * 512
  int f = tid >> 9, pos = tid & 511;
  int kt = f >> 3, nt = f & 7;
  int half = pos >> 8, lane = (pos >> 3) & 31, io = pos & 7;
  int i = half * 8 + io;
  int K = ((lane >> 4) << 4) + i;
  int jrow = kt * 32 + K;
  int l = nt * 16 + (lane & 15);
  ws[W3P_OFF + (size_t)f * FRAG + pos] = (_Float16)W3[(size_t)jrow * HH + l];
}

#define WMMA(a, b, c) \
  __builtin_amdgcn_wmma_f32_16x16x32_f16(false, (a), false, (b), (short)0, (c), false, false)

__global__ void __launch_bounds__(256, 1)
gru_main(const _Float16* __restrict__ ws, const float* __restrict__ b1,
         const float* __restrict__ b2, float* __restrict__ out) {
  // [parity][W][row(=2j x 8kb)][ntpair][512 halfs] = 128 KB, TDM double buffer
  __shared__ __align__(16) _Float16 lbuf[2][2][16][2][FRAG];

  const _Float16* x16 = ws + X16_OFF;
  const _Float16* s16 = ws + S16_OFF;
  const _Float16* w3p = ws + W3P_OFF;

  int tid  = threadIdx.x;
  int lane = tid & 31, wv = tid >> 5;
  int bg = blockIdx.x & 31;                 // 32 groups of 8 b-tiles
  int lt = blockIdx.x >> 5;                 // 4 l-macrotiles of 32 cols
  int btile = bg * 8 + wv;                  // 0..255, 16 rows each
  int bBase = btile * 16;
  int nt0 = lt * 2;
  int l15 = lane & 15, h = lane >> 4;
  int bA = bBase + l15;

  // Make LDS "written" in the compiler's view (TDM writes are invisible to it),
  // and give deterministic contents before the first DMA.
  {
    uint32_t* lb = (uint32_t*)&lbuf[0][0][0][0][0];
#pragma unroll
    for (int i = 0; i < 128; ++i) lb[tid + 256 * i] = (uint32_t)tid;
  }
  __syncthreads();

  // ---- TDM 2D descriptor: 16 rows x 2KB, row stride 8KB, dword elements ----
  const i32x8 g1 = { 0x00020000,   // data_size = 4B
                     0x02000000,   // tensor_dim0 = 512 (dwords) << 16
                     0x00100000,   // tensor_dim1 = 16 << 16
                     0x02000000,   // tile_dim0 = 512 << 16
                     0x00000010,   // tile_dim1 = 16
                     2048,         // tensor_dim0_stride = 2048 dwords (8KB)
                     0, 0 };
  const i32x4 zz = { 0, 0, 0, 0 };
  const i32x8 z8 = { 0, 0, 0, 0, 0, 0, 0, 0 };
  const uint64_t gbase1 =
      (uint64_t)(uintptr_t)(ws + W1P_OFF) + (uint64_t)nt0 * 1024u;
  const uint64_t gbase2 =
      (uint64_t)(uintptr_t)(ws + W2P_OFF) + (uint64_t)nt0 * 1024u;

  auto issue = [&](int par, int js) {                // js even: 2 j-steps
    uint64_t a1 = gbase1 + (uint64_t)js * 65536u;    // j * 64 frags * 1KB
    uint64_t a2 = gbase2 + (uint64_t)js * 65536u;
    unsigned lds1 = (unsigned)(uintptr_t)&lbuf[par][0][0][0][0];
    unsigned lds2 = (unsigned)(uintptr_t)&lbuf[par][1][0][0][0];
    u32x4 ga;
    ga.x = 1u;                                       // count = 1
    ga.y = lds1;
    ga.z = (unsigned)a1;
    ga.w = (unsigned)(a1 >> 32) | 0x80000000u;       // type = 2 ("image")
    __builtin_amdgcn_tensor_load_to_lds(ga, g1, zz, zz, z8, 0);
    ga.y = lds2;
    ga.z = (unsigned)a2;
    ga.w = (unsigned)(a2 >> 32) | 0x80000000u;
    __builtin_amdgcn_tensor_load_to_lds(ga, g1, zz, zz, z8, 0);
  };

  if (wv == 0) issue(0, 0);

  // Preload per-lane x-row chunks for the 8 k-subblocks of a j-row (64 VGPRs).
  v16h xA[8];
#pragma unroll
  for (int kb = 0; kb < 8; ++kb)
    xA[kb] = load_arow(x16 + (size_t)bA * TWOH + kb * 32, h);

  v8f m1[2] = {{0.f}, {0.f}};
  v8f m2[2] = {{0.f}, {0.f}};
  const _Float16* sArow = s16 + (size_t)bA * TWOH;

  if (wv == 0) __builtin_amdgcn_s_wait_tensorcnt(0);
  __syncthreads();

  for (int js = 0; js < TWOH; js += 2) {
    int par = (js >> 1) & 1;
    if (wv == 0 && js + 2 < TWOH) issue(par ^ 1, js + 2);

    _Float16 s0 = sArow[js], s1 = sArow[js + 1];
    v16h sA0, sA1;
#pragma unroll
    for (int i = 0; i < 16; ++i) { sA0[i] = s0; sA1[i] = s1; }

    // Software-pipelined over the 16 rows of this stage: load row r+1's
    // fragments into bn[] while the 4 WMMAs consume bc[] (distinct regs).
    v16h bc[4], bn[4];
    bc[0] = load_frag(&lbuf[par][0][0][0][0], lane);
    bc[1] = load_frag(&lbuf[par][0][0][1][0], lane);
    bc[2] = load_frag(&lbuf[par][1][0][0][0], lane);
    bc[3] = load_frag(&lbuf[par][1][0][1][0], lane);
#pragma unroll
    for (int r = 0; r < 16; ++r) {
      if (r < 15) {
        bn[0] = load_frag(&lbuf[par][0][r + 1][0][0], lane);
        bn[1] = load_frag(&lbuf[par][0][r + 1][1][0], lane);
        bn[2] = load_frag(&lbuf[par][1][r + 1][0][0], lane);
        bn[3] = load_frag(&lbuf[par][1][r + 1][1][0], lane);
      }
      v16h a = (r < 8 ? sA0 : sA1) * xA[r & 7];      // rank-1 A tile
      m1[0] = WMMA(a, bc[0], m1[0]);
      m1[1] = WMMA(a, bc[1], m1[1]);
      m2[0] = WMMA(a, bc[2], m2[0]);
      m2[1] = WMMA(a, bc[3], m2[1]);
#pragma unroll
      for (int q = 0; q < 4; ++q) bc[q] = bn[q];
    }

    if (wv == 0) __builtin_amdgcn_s_wait_tensorcnt(0);
    __syncthreads();
  }

  // s @ W3 via WMMA (K = 256 -> 8 steps), W3 fragments straight from L2.
  v8f sw[2] = {{0.f}, {0.f}};
#pragma unroll
  for (int kt = 0; kt < 8; ++kt) {
    v16h aA  = load_arow(sArow + kt * 32, h);
    v16h w30 = load_frag(w3p + ((size_t)kt * NFN + nt0) * FRAG, lane);
    v16h w31 = load_frag(w3p + ((size_t)kt * NFN + nt0 + 1) * FRAG, lane);
    sw[0] = WMMA(aA, w30, sw[0]);
    sw[1] = WMMA(aA, w31, sw[1]);
  }

  // Epilogue: biases + sigmoid/tanh blend, coalesced f32 stores.
  int brow0 = bBase + 8 * h;                  // C/D layout: M = r + 8*(lane>=16)
#pragma unroll
  for (int lt2 = 0; lt2 < 2; ++lt2) {
    int l = lt * 32 + lt2 * 16 + l15;
    float bb1 = b1[l], bb2 = b2[l];
#pragma unroll
    for (int r = 0; r < 8; ++r) {
      float u  = 1.0f / (1.0f + __expf(-(m2[lt2][r] + bb2)));
      float st = tanhf(m1[lt2][r] + bb1);
      float o  = u * st + (1.0f - u) * sw[lt2][r];
      out[(size_t)(brow0 + r) * HH + l] = o;
    }
  }
}

extern "C" void kernel_launch(void* const* d_in, const int* in_sizes, int n_in,
                              void* d_out, int out_size, void* d_ws, size_t ws_size,
                              hipStream_t stream) {
  const float* in0 = (const float*)d_in[0];
  const float* in1 = (const float*)d_in[1];
  const float* st0 = (const float*)d_in[2];
  const float* st1 = (const float*)d_in[3];
  const float* W1  = (const float*)d_in[4];
  const float* b1  = (const float*)d_in[5];
  const float* W2  = (const float*)d_in[6];
  const float* b2  = (const float*)d_in[7];
  const float* W3  = (const float*)d_in[8];
  _Float16* ws = (_Float16*)d_ws;
  float* out = (float*)d_out;

  pack_sx<<<BSZ, 256, 0, stream>>>(in0, in1, st0, st1, ws);
  pack_W<<<65536, 256, 0, stream>>>(W1, W2, ws);
  pack_W3<<<128, 256, 0, stream>>>(W3, ws);
  gru_main<<<128, 256, 0, stream>>>(ws, b1, b2, out);   // 1024 wave-tiles
}
